// TransformerBlock_83708912599211
// MI455X (gfx1250) — matrix operand order repaired and checked by reading, not compiled, not run on hardware
//
#include <hip/hip_runtime.h>
#include <math.h>

// ---------------- types ----------------
typedef __attribute__((ext_vector_type(16))) __bf16   bf16x16;
typedef __attribute__((ext_vector_type(8)))  float    f32x8;
typedef __attribute__((ext_vector_type(4)))  unsigned int u32x4;

union Frag { bf16x16 v; u32x4 q[2]; };

__device__ __forceinline__ unsigned short f32_to_bf16(float f) {
    unsigned int u = __float_as_uint(f);
    u += 0x7fffu + ((u >> 16) & 1u);
    return (unsigned short)(u >> 16);
}

// Swin constants
#define TB   32
#define TC   256
#define TH   56
#define TW   56
#define TM   7
#define TN   49          // tokens per window
#define NWIN 2048        // 32 * 8 * 8
#define TT   100352      // NWIN * 49  (multiple of 256)
#define NHEADS 8
#define DHEAD  32

#define LDSB_STRIDE 136   // 64x128 chunk padded: 272B row = 68 banks (conflict-free-ish)
#define LDSB_ELEMS  (64 * LDSB_STRIDE)
#define ATT_STRIDE  72    // 144B row = 36 banks

// windowed token t, channel c -> flat index into [B,C,H,W]
__device__ __forceinline__ size_t xp_index(int t, int c) {
    int wdx = t / TN, p = t % TN;
    int b  = wdx >> 6, wi = wdx & 63;
    int wh = wi >> 3,  ww = wi & 7;
    int mh = p / TM,   mw = p % TM;
    int hh = wh * TM + mh, wc = ww * TM + mw;
    return (((size_t)b * TC + c) * TH + hh) * TW + wc;
}

// ---------------- small prep kernels ----------------
__global__ void k_cvt_bf16(const float* __restrict__ src,
                           unsigned short* __restrict__ dst, int n) {
    int i = blockIdx.x * 256 + threadIdx.x;
    if (i < n) dst[i] = f32_to_bf16(src[i]);
}

__global__ void k_bias_expand(const float* __restrict__ table,
                              float* __restrict__ bias49) {
    int i = blockIdx.x * 256 + threadIdx.x;
    if (i >= NHEADS * TN * TN) return;
    int h   = i / (TN * TN);
    int rem = i % (TN * TN);
    int a = rem / TN, b = rem % TN;
    int ih = a / TM, iw = a % TM, jh = b / TM, jw = b % TM;
    int idx = (ih - jh + (TM - 1)) * (2 * TM - 1) + (iw - jw + (TM - 1));
    bias49[i] = table[idx * NHEADS + h];
}

// block-wide sum of one value per thread (256 thr, wave32): shfl + one LDS round
__device__ __forceinline__ float block_sum256(float v, float* red8) {
    #pragma unroll
    for (int off = 1; off < 32; off <<= 1) v += __shfl_xor(v, off, 32);
    int wave = threadIdx.x >> 5;
    if ((threadIdx.x & 31) == 0) red8[wave] = v;
    __syncthreads();
    float tot = red8[0] + red8[1] + red8[2] + red8[3] +
                red8[4] + red8[5] + red8[6] + red8[7];
    __syncthreads();
    return tot;
}

// LN over windowed x -> bf16 tokens [T, 256]
__global__ void k_ln1(const float* __restrict__ x,
                      const float* __restrict__ g, const float* __restrict__ b,
                      unsigned short* __restrict__ h1) {
    __shared__ float red8[8];
    int t = blockIdx.x;
    int c = threadIdx.x;
    float v = x[xp_index(t, c)];
    float mu = block_sum256(v, red8) * (1.0f / 256.0f);
    float d = v - mu;
    float var = block_sum256(d * d, red8) * (1.0f / 256.0f);
    float rstd = rsqrtf(var + 1e-5f);
    h1[(size_t)t * TC + c] = f32_to_bf16(d * rstd * g[c] + b[c]);
}

// LN over token-major fp32 x2 -> bf16 tokens [T, 256]
__global__ void k_ln2(const float* __restrict__ x2,
                      const float* __restrict__ g, const float* __restrict__ b,
                      unsigned short* __restrict__ hb) {
    __shared__ float red8[8];
    int t = blockIdx.x;
    int c = threadIdx.x;
    float v = x2[(size_t)t * TC + c];
    float mu = block_sum256(v, red8) * (1.0f / 256.0f);
    float d = v - mu;
    float var = block_sum256(d * d, red8) * (1.0f / 256.0f);
    float rstd = rsqrtf(var + 1e-5f);
    hb[(size_t)t * TC + c] = f32_to_bf16(d * rstd * g[c] + b[c]);
}

// ---------------- WMMA GEMM mainloop ----------------
// C[row,col] = sum_k A[row,k]*W[col,k]; A:[T,K] bf16, W:[N,K] bf16 row-major.
// Block: 256 threads (8 waves). Wave w: 32 rows x 64 cols (2 row-tiles x 4 col-tiles).
// Block tile: 256 rows x 64 cols. B panel double-buffered in LDS, 64x128 K-chunks.
__device__ __forceinline__ void gemm_mainloop(
    const unsigned short* __restrict__ A,
    const unsigned short* __restrict__ W,
    int K, int rowbase, int colbase,
    unsigned short* ldsB,            // 2 * LDSB_ELEMS
    f32x8 acc[2][4]) {
    const int tid  = threadIdx.x;
    const int lane = tid & 31;
    const int lg   = lane >> 4;   // lane group 0/1
    const int lr   = lane & 15;

    f32x8 z = {};
    #pragma unroll
    for (int rt = 0; rt < 2; ++rt)
        #pragma unroll
        for (int jt = 0; jt < 4; ++jt) acc[rt][jt] = z;

    const unsigned short* arow0 = A + (size_t)(rowbase + lr) * K;
    const unsigned short* arow1 = A + (size_t)(rowbase + 16 + lr) * K;

    // staging chunk geometry: 1024 16B-chunks, 4 per thread
    int sn[4], sk[4];
    #pragma unroll
    for (int i = 0; i < 4; ++i) {
        int qi = tid + 256 * i;
        sn[i] = qi >> 4;            // 0..63 (col within panel)
        sk[i] = (qi & 15) << 3;     // 0..120 (k within chunk)
    }

    // prologue: stage chunk 0
    u32x4 stage[4];
    #pragma unroll
    for (int i = 0; i < 4; ++i)
        stage[i] = *(const u32x4*)(W + (size_t)(colbase + sn[i]) * K + sk[i]);
    #pragma unroll
    for (int i = 0; i < 4; ++i)
        *(u32x4*)(ldsB + sn[i] * LDSB_STRIDE + sk[i]) = stage[i];
    __syncthreads();

    for (int kb = 0; kb < K; kb += 128) {
        const int cur = (kb >> 7) & 1;
        unsigned short* buf = ldsB + cur * LDSB_ELEMS;
        const bool more = (kb + 128) < K;

        if (more) {
            // issue next-chunk global loads; they complete during WMMA work below
            #pragma unroll
            for (int i = 0; i < 4; ++i)
                stage[i] = *(const u32x4*)(W + (size_t)(colbase + sn[i]) * K + (kb + 128) + sk[i]);
            __builtin_prefetch(arow0 + kb + 128, 0, 1);   // global_prefetch_b8
        }

        #pragma unroll
        for (int ks = 0; ks < 128; ks += 32) {
            // A fragments: lane row lr, K chunks lg*8..+7 and 16+lg*8..+7
            Frag a0, a1;
            const unsigned short* ap0 = arow0 + kb + ks + (lg << 3);
            const unsigned short* ap1 = arow1 + kb + ks + (lg << 3);
            a0.q[0] = *(const u32x4*)(ap0);
            a0.q[1] = *(const u32x4*)(ap0 + 16);
            a1.q[0] = *(const u32x4*)(ap1);
            a1.q[1] = *(const u32x4*)(ap1 + 16);
            #pragma unroll
            for (int jt = 0; jt < 4; ++jt) {
                // B fragment: lane col lr, K chunks lg*8..+7 and 16+lg*8..+7
                Frag bf;
                const unsigned short* bp = buf + (jt * 16 + lr) * LDSB_STRIDE + ks + (lg << 3);
                bf.q[0] = *(const u32x4*)(bp);
                bf.q[1] = *(const u32x4*)(bp + 16);
                acc[0][jt] = __builtin_amdgcn_wmma_f32_16x16x32_bf16(
                    false, a0.v, false, bf.v, (short)0, acc[0][jt], false, false);
                acc[1][jt] = __builtin_amdgcn_wmma_f32_16x16x32_bf16(
                    false, a1.v, false, bf.v, (short)0, acc[1][jt], false, false);
            }
        }

        if (more) {
            unsigned short* nbuf = ldsB + (cur ^ 1) * LDSB_ELEMS;
            #pragma unroll
            for (int i = 0; i < 4; ++i)
                *(u32x4*)(nbuf + sn[i] * LDSB_STRIDE + sk[i]) = stage[i];
            __syncthreads();
        }
    }
}

// ---------------- GEMM kernels (different epilogues) ----------------
__global__ void k_gemm_qkv(const unsigned short* __restrict__ A,
                           const unsigned short* __restrict__ W,
                           const float* __restrict__ bias,
                           unsigned short* __restrict__ out) {
    __shared__ unsigned short ldsB[2 * LDSB_ELEMS];
    f32x8 acc[2][4];
    int rowbase = blockIdx.x * 256 + (threadIdx.x >> 5) * 32;
    int colbase = blockIdx.y * 64;
    gemm_mainloop(A, W, 256, rowbase, colbase, ldsB, acc);
    int lane = threadIdx.x & 31, lg = lane >> 4, lr = lane & 15;
    const float qscale = 0.35355339059327379f;   // HEADS**-0.5
    #pragma unroll
    for (int jt = 0; jt < 4; ++jt) {
        int col = colbase + jt * 16 + lr;
        float bb = bias[col];
        float sc = (col < 256) ? qscale : 1.0f;
        #pragma unroll
        for (int rt = 0; rt < 2; ++rt)
            #pragma unroll
            for (int r = 0; r < 8; ++r) {
                int row = rowbase + rt * 16 + r + (lg << 3);
                out[(size_t)row * 768 + col] = f32_to_bf16((acc[rt][jt][r] + bb) * sc);
            }
    }
}

__global__ void k_gemm_proj(const unsigned short* __restrict__ A,
                            const unsigned short* __restrict__ W,
                            const float* __restrict__ bias,
                            const float* __restrict__ x,
                            float* __restrict__ x2) {
    __shared__ unsigned short ldsB[2 * LDSB_ELEMS];
    f32x8 acc[2][4];
    int rowbase = blockIdx.x * 256 + (threadIdx.x >> 5) * 32;
    int colbase = blockIdx.y * 64;
    gemm_mainloop(A, W, 256, rowbase, colbase, ldsB, acc);
    int lane = threadIdx.x & 31, lg = lane >> 4, lr = lane & 15;
    #pragma unroll
    for (int jt = 0; jt < 4; ++jt) {
        int col = colbase + jt * 16 + lr;
        float bb = bias[col];
        #pragma unroll
        for (int rt = 0; rt < 2; ++rt)
            #pragma unroll
            for (int r = 0; r < 8; ++r) {
                int row = rowbase + rt * 16 + r + (lg << 3);
                float res = x[xp_index(row, col)];
                x2[(size_t)row * TC + col] = acc[rt][jt][r] + bb + res;
            }
    }
}

__global__ void k_gemm_fc1(const unsigned short* __restrict__ A,
                           const unsigned short* __restrict__ W,
                           const float* __restrict__ bias,
                           unsigned short* __restrict__ out) {
    __shared__ unsigned short ldsB[2 * LDSB_ELEMS];
    f32x8 acc[2][4];
    int rowbase = blockIdx.x * 256 + (threadIdx.x >> 5) * 32;
    int colbase = blockIdx.y * 64;
    gemm_mainloop(A, W, 256, rowbase, colbase, ldsB, acc);
    int lane = threadIdx.x & 31, lg = lane >> 4, lr = lane & 15;
    #pragma unroll
    for (int jt = 0; jt < 4; ++jt) {
        int col = colbase + jt * 16 + lr;
        float bb = bias[col];
        #pragma unroll
        for (int rt = 0; rt < 2; ++rt)
            #pragma unroll
            for (int r = 0; r < 8; ++r) {
                int row = rowbase + rt * 16 + r + (lg << 3);
                float v = acc[rt][jt][r] + bb;
                v = 0.5f * v * (1.0f + erff(v * 0.70710678118654752f));  // exact GELU
                out[(size_t)row * 1024 + col] = f32_to_bf16(v);
            }
    }
}

__global__ void k_gemm_fc2(const unsigned short* __restrict__ A,
                           const unsigned short* __restrict__ W,
                           const float* __restrict__ bias,
                           const float* __restrict__ x2,
                           float* __restrict__ y) {
    __shared__ unsigned short ldsB[2 * LDSB_ELEMS];
    f32x8 acc[2][4];
    int rowbase = blockIdx.x * 256 + (threadIdx.x >> 5) * 32;
    int colbase = blockIdx.y * 64;
    gemm_mainloop(A, W, 1024, rowbase, colbase, ldsB, acc);
    int lane = threadIdx.x & 31, lg = lane >> 4, lr = lane & 15;
    #pragma unroll
    for (int jt = 0; jt < 4; ++jt) {
        int col = colbase + jt * 16 + lr;
        float bb = bias[col];
        #pragma unroll
        for (int rt = 0; rt < 2; ++rt)
            #pragma unroll
            for (int r = 0; r < 8; ++r) {
                int row = rowbase + rt * 16 + r + (lg << 3);
                float v = acc[rt][jt][r] + bb + x2[(size_t)row * TC + col];
                y[xp_index(row, col)] = v;   // window reverse
            }
    }
}

// ---------------- windowed attention ----------------
// One block per window (256 thr = 8 waves). Wave h handles head h.
__global__ void k_attn(const unsigned short* __restrict__ qkv,
                       const float* __restrict__ bias49,
                       unsigned short* __restrict__ attnout) {
    __shared__ unsigned short sh[8 * (32 + 16) * ATT_STRIDE];  // per wave: V^T 32 rows + P 16 rows
    int win  = blockIdx.x;
    int h    = threadIdx.x >> 5;
    int lane = threadIdx.x & 31;
    int lg = lane >> 4, lr = lane & 15;
    size_t t0 = (size_t)win * TN;

    const unsigned short* qb = qkv + t0 * 768 + h * DHEAD;
    const unsigned short* kb = qb + 256;
    const unsigned short* vb = qb + 512;
    unsigned short* svt = sh + h * 48 * ATT_STRIDE;     // V^T: [d=32][j padded]
    unsigned short* sp  = svt + 32 * ATT_STRIDE;        // P  : [m=16][j padded]

    // stage V^T (zero-pad j in 49..63)
    for (int idx = lane; idx < 32 * 64; idx += 32) {
        int d = idx >> 6, j = idx & 63;
        unsigned short vv = 0;
        if (j < TN) vv = vb[(size_t)j * 768 + d];
        svt[d * ATT_STRIDE + j] = vv;
    }
    asm volatile("s_wait_dscnt 0x0" ::: "memory");

    const float* bh = bias49 + h * (TN * TN);
    u32x4 zq = {0u, 0u, 0u, 0u};

    for (int mt = 0; mt < 4; ++mt) {
        int m0 = mt * 16;
        // A fragment of q (rows >= 49 zero)
        Frag aq;
        {
            int rp = m0 + lr;
            if (rp < TN) {
                const unsigned short* p = qb + (size_t)rp * 768 + (lg << 3);
                aq.q[0] = *(const u32x4*)(p);
                aq.q[1] = *(const u32x4*)(p + 16);
            } else { aq.q[0] = zq; aq.q[1] = zq; }
        }
        // S tiles
        f32x8 s[4];
        #pragma unroll
        for (int jt = 0; jt < 4; ++jt) {
            Frag bk;
            int jp = jt * 16 + lr;
            if (jp < TN) {
                const unsigned short* p = kb + (size_t)jp * 768 + (lg << 3);
                bk.q[0] = *(const u32x4*)(p);
                bk.q[1] = *(const u32x4*)(p + 16);
            } else { bk.q[0] = zq; bk.q[1] = zq; }
            f32x8 z = {};
            s[jt] = __builtin_amdgcn_wmma_f32_16x16x32_bf16(
                false, aq.v, false, bk.v, (short)0, z, false, false);
        }
        // bias + mask + softmax per row
        #pragma unroll
        for (int r = 0; r < 8; ++r) {
            int rp = m0 + r + (lg << 3);
            float mx = -3.0e38f;
            #pragma unroll
            for (int jt = 0; jt < 4; ++jt) {
                int jg = jt * 16 + lr;
                float v;
                if (rp < TN && jg < TN) v = s[jt][r] + bh[rp * TN + jg];
                else                    v = -3.0e38f;
                s[jt][r] = v;
                mx = fmaxf(mx, v);
            }
            #pragma unroll
            for (int off = 1; off < 16; off <<= 1)
                mx = fmaxf(mx, __shfl_xor(mx, off, 16));
            float sum = 0.0f;
            #pragma unroll
            for (int jt = 0; jt < 4; ++jt) {
                float e = __expf(s[jt][r] - mx);
                s[jt][r] = e;
                sum += e;
            }
            #pragma unroll
            for (int off = 1; off < 16; off <<= 1)
                sum += __shfl_xor(sum, off, 16);
            float inv = 1.0f / sum;
            #pragma unroll
            for (int jt = 0; jt < 4; ++jt) s[jt][r] *= inv;
        }
        // store P strip to LDS (bf16)
        #pragma unroll
        for (int r = 0; r < 8; ++r) {
            int m = r + (lg << 3);
            #pragma unroll
            for (int jt = 0; jt < 4; ++jt)
                sp[m * ATT_STRIDE + jt * 16 + lr] = f32_to_bf16(s[jt][r]);
        }
        asm volatile("s_wait_dscnt 0x0" ::: "memory");
        // O = P * V  (16x32, K=64 -> 2 k-steps, 2 col tiles)
        f32x8 o0 = {}, o1 = {};
        #pragma unroll
        for (int kk = 0; kk < 64; kk += 32) {
            Frag ap;
            const unsigned short* pp = sp + lr * ATT_STRIDE + kk + (lg << 4);
            ap.q[0] = *(const u32x4*)(pp);
            ap.q[1] = *(const u32x4*)(pp + 8);
            Frag bv0, bv1;
            const unsigned short* vp0 = svt + (0 * 16 + lr) * ATT_STRIDE + kk + (lg << 4);
            const unsigned short* vp1 = svt + (1 * 16 + lr) * ATT_STRIDE + kk + (lg << 4);
            bv0.q[0] = *(const u32x4*)(vp0); bv0.q[1] = *(const u32x4*)(vp0 + 8);
            bv1.q[0] = *(const u32x4*)(vp1); bv1.q[1] = *(const u32x4*)(vp1 + 8);
            o0 = __builtin_amdgcn_wmma_f32_16x16x32_bf16(
                false, ap.v, false, bv0.v, (short)0, o0, false, false);
            o1 = __builtin_amdgcn_wmma_f32_16x16x32_bf16(
                false, ap.v, false, bv1.v, (short)0, o1, false, false);
        }
        // write O rows < 49
        #pragma unroll
        for (int r = 0; r < 8; ++r) {
            int rp = m0 + r + (lg << 3);
            if (rp < TN) {
                attnout[(t0 + rp) * 256 + h * DHEAD + lr]      = f32_to_bf16(o0[r]);
                attnout[(t0 + rp) * 256 + h * DHEAD + 16 + lr] = f32_to_bf16(o1[r]);
            }
        }
        asm volatile("s_wait_dscnt 0x0" ::: "memory");  // P strip reuse next iter
    }
}

// ---------------- launcher ----------------
extern "C" void kernel_launch(void* const* d_in, const int* in_sizes, int n_in,
                              void* d_out, int out_size, void* d_ws, size_t ws_size,
                              hipStream_t stream) {
    const float* x      = (const float*)d_in[0];
    const float* qkv_w  = (const float*)d_in[1];
    const float* qkv_b  = (const float*)d_in[2];
    const float* proj_w = (const float*)d_in[3];
    const float* proj_b = (const float*)d_in[4];
    const float* btab   = (const float*)d_in[5];
    const float* ln1_g  = (const float*)d_in[6];
    const float* ln1_b  = (const float*)d_in[7];
    const float* ln2_g  = (const float*)d_in[8];
    const float* ln2_b  = (const float*)d_in[9];
    const float* w1     = (const float*)d_in[10];
    const float* b1     = (const float*)d_in[11];
    const float* w2     = (const float*)d_in[12];
    const float* b2     = (const float*)d_in[13];
    float* out = (float*)d_out;

    char* ws = (char*)d_ws;
    // 256-aligned workspace offsets (bytes)
    unsigned short* Wqkv   = (unsigned short*)(ws + 0);            // 768*256*2
    unsigned short* Wproj  = (unsigned short*)(ws + 393216);       // 256*256*2
    unsigned short* W1     = (unsigned short*)(ws + 524288);       // 1024*256*2
    unsigned short* W2     = (unsigned short*)(ws + 1048576);      // 256*1024*2
    float*          bias49 = (float*)        (ws + 1572864);       // 8*49*49*4
    unsigned short* h1     = (unsigned short*)(ws + 1649920);      // T*256*2
    unsigned short* qkvbuf = (unsigned short*)(ws + 53030144);     // T*768*2
    unsigned short* attbuf = (unsigned short*)(ws + 207170816);    // T*256*2
    float*          x2     = (float*)        (ws + 258551040);     // T*256*4
    unsigned short* h2buf  = (unsigned short*)(ws + 361311488);    // T*1024*2
    // total: 566,832,384 bytes

    // weight conversion to bf16
    k_cvt_bf16<<<(768 * 256 + 255) / 256, 256, 0, stream>>>(qkv_w, Wqkv, 768 * 256);
    k_cvt_bf16<<<(256 * 256 + 255) / 256, 256, 0, stream>>>(proj_w, Wproj, 256 * 256);
    k_cvt_bf16<<<(1024 * 256 + 255) / 256, 256, 0, stream>>>(w1, W1, 1024 * 256);
    k_cvt_bf16<<<(256 * 1024 + 255) / 256, 256, 0, stream>>>(w2, W2, 256 * 1024);
    k_bias_expand<<<(NHEADS * TN * TN + 255) / 256, 256, 0, stream>>>(btab, bias49);

    // LN1 (window partition fused)
    k_ln1<<<TT, 256, 0, stream>>>(x, ln1_g, ln1_b, h1);
    // QKV GEMM [T,256]x[768,256]^T
    k_gemm_qkv<<<dim3(TT / 256, 768 / 64), 256, 0, stream>>>(h1, Wqkv, qkv_b, qkvbuf);
    // windowed attention
    k_attn<<<NWIN, 256, 0, stream>>>(qkvbuf, bias49, attbuf);
    // proj GEMM + residual (x windowed)
    k_gemm_proj<<<dim3(TT / 256, 256 / 64), 256, 0, stream>>>(attbuf, Wproj, proj_b, x, x2);
    // LN2 (reuse h1)
    k_ln2<<<TT, 256, 0, stream>>>(x2, ln2_g, ln2_b, h1);
    // MLP
    k_gemm_fc1<<<dim3(TT / 256, 1024 / 64), 256, 0, stream>>>(h1, W1, b1, h2buf);
    k_gemm_fc2<<<dim3(TT / 256, 256 / 64), 256, 0, stream>>>(h2buf, W2, b2, x2, out);
}
